// EnhanceLayer_79499844649022
// MI455X (gfx1250) — matrix operand order repaired and checked by reading, not compiled, not run on hardware
//
#include <hip/hip_runtime.h>
#include <hip/hip_bf16.h>

typedef __attribute__((ext_vector_type(16))) __bf16 v16bf;
typedef __attribute__((ext_vector_type(8)))  float  v8f;

#define IN_DIM 64

__device__ __forceinline__ v8f wmma_bf16(v16bf a, v16bf b, v8f c) {
    return __builtin_amdgcn_wmma_f32_16x16x32_bf16(false, a, false, b, (short)0, c, false, false);
}

// raw v_sqrt_f32 (~1 ulp) -- avoids the correctly-rounded software expansion
__device__ __forceinline__ float fast_sqrt(float x) { return __builtin_amdgcn_sqrtf(x); }

// one butterfly step: x += lane(x ^ XORM), via ds_swizzle group-of-32 immediate pattern
template <int XORM>
__device__ __forceinline__ float swz_add(float x) {
    int y = __builtin_amdgcn_ds_swizzle(__float_as_int(x), (XORM << 10) | 0x1f);
    return x + __int_as_float(y);
}
// sum of x over each 16-lane half (masks < 16 never cross the halves)
__device__ __forceinline__ float hsum16(float x) {
    x = swz_add<1>(x); x = swz_add<2>(x); x = swz_add<4>(x); x = swz_add<8>(x);
    return x;
}

// order-preserving float->uint map for atomic max on floats
__device__ __forceinline__ unsigned f2o(float f) {
    unsigned u = __float_as_uint(f);
    return (u & 0x80000000u) ? ~u : (u | 0x80000000u);
}
__device__ __forceinline__ float o2f(unsigned u) {
    return (u & 0x80000000u) ? __uint_as_float(u & 0x7fffffffu) : __uint_as_float(~u);
}
#define ENC_NEG_INF 0x007FFFFFu   // f2o(-inf)

// load 8 contiguous f32 -> bf16 into A fragment positions [pos..pos+7]
__device__ __forceinline__ void fill8(v16bf& A, int pos, const float* p, int k) {
    float4 a = *(const float4*)(p + k);
    float4 b = *(const float4*)(p + k + 4);
    A[pos+0]=(__bf16)a.x; A[pos+1]=(__bf16)a.y; A[pos+2]=(__bf16)a.z; A[pos+3]=(__bf16)a.w;
    A[pos+4]=(__bf16)b.x; A[pos+5]=(__bf16)b.y; A[pos+6]=(__bf16)b.z; A[pos+7]=(__bf16)b.w;
}
__device__ __forceinline__ void fill8_diff(v16bf& A, int pos, const float* ps, const float* pd, int k) {
    float4 s0=*(const float4*)(ps+k), s1=*(const float4*)(ps+k+4);
    float4 d0=*(const float4*)(pd+k), d1=*(const float4*)(pd+k+4);
    A[pos+0]=(__bf16)fmaxf(s0.x-d0.x,0.f); A[pos+1]=(__bf16)fmaxf(s0.y-d0.y,0.f);
    A[pos+2]=(__bf16)fmaxf(s0.z-d0.z,0.f); A[pos+3]=(__bf16)fmaxf(s0.w-d0.w,0.f);
    A[pos+4]=(__bf16)fmaxf(s1.x-d1.x,0.f); A[pos+5]=(__bf16)fmaxf(s1.y-d1.y,0.f);
    A[pos+6]=(__bf16)fmaxf(s1.z-d1.z,0.f); A[pos+7]=(__bf16)fmaxf(s1.w-d1.w,0.f);
}

// ---------------- Kernel 1: Q_h = h @ W_Q (bf16 WMMA, f32 accum)
//  + per-(row, head) L2 norms of Q_h  + init d_out to enc(-inf)
__global__ void __launch_bounds__(32)
proj_q_kernel(const float* __restrict__ h, const float* __restrict__ Wq,
              float* __restrict__ Qh, float* __restrict__ Qn,
              unsigned* __restrict__ outEnc) {
    __shared__ __align__(32) __bf16 sB[2][4][32][16];   // [kstep][ntile][lane][elem] 8KB
    int lane = threadIdx.x;
    for (int s = lane; s < 256; s += 32) {
        int kstep = s >> 7, ntile = (s >> 5) & 3, sl = s & 31;
        int n  = ntile*16 + (sl & 15);
        int kh = sl >> 4;
        #pragma unroll
        for (int i = 0; i < 16; ++i)
            sB[kstep][ntile][sl][i] = (__bf16)Wq[(kstep*32 + kh*16 + i)*IN_DIM + n];
    }
    __syncthreads();

    int tile  = blockIdx.x;
    int lrow  = lane & 15, khalf = lane >> 4;
    const float* rp = h + (size_t)(tile*16 + lrow) * IN_DIM;
    v16bf A[2];
    #pragma unroll
    for (int ks = 0; ks < 2; ++ks) {
        int kb = ks*32 + khalf*16;
        fill8(A[ks], 0, rp, kb);
        fill8(A[ks], 8, rp, kb + 8);
    }
    int n = lane & 15, mh = lane >> 4;
    #pragma unroll
    for (int nt = 0; nt < 4; ++nt) {
        v8f c = {};
        #pragma unroll
        for (int ks = 0; ks < 2; ++ks) {
            v16bf b = *(const v16bf*)&sB[ks][nt][lane][0];
            c = wmma_bf16(A[ks], b, c);
        }
        #pragma unroll
        for (int j = 0; j < 8; ++j) {
            int row = tile*16 + mh*8 + j;
            size_t idx = (size_t)row*IN_DIM + nt*16 + n;
            Qh[idx]     = c[j];
            outEnc[idx] = ENC_NEG_INF;
            float sq = hsum16(c[j]*c[j]);          // ||Q_h[row, head]||^2
            if (n == 0) Qn[(size_t)row*4 + nt] = fast_sqrt(sq);
        }
    }
}

// ---------------- Kernel 2: fused edge pipeline
// rows = flattened [E*2, 64]; 8 waves/block, 16 rows/wave, grid exact (full EXEC for WMMA)
__global__ void __launch_bounds__(256)
edge_kernel(const float* __restrict__ h, const float* __restrict__ e,
            const float* __restrict__ Wk, const float* __restrict__ Wv,
            const float* __restrict__ Wr, const int* __restrict__ src,
            const int* __restrict__ dst, const float* __restrict__ Qh,
            const float* __restrict__ Qn, unsigned* __restrict__ outEnc) {
    __shared__ __align__(32) __bf16 sB[3][2][4][32][16];   // 24KB, fragment-order weights
    int t = threadIdx.x;
    {
        int kstep = t >> 7, ntile = (t >> 5) & 3, sl = t & 31;
        int n  = ntile*16 + (sl & 15);
        int kh = sl >> 4;
        const float* Ws[3] = {Wk, Wv, Wr};
        #pragma unroll
        for (int m = 0; m < 3; ++m)
            #pragma unroll
            for (int i = 0; i < 16; ++i)
                sB[m][kstep][ntile][sl][i] = (__bf16)Ws[m][(kstep*32 + kh*16 + i)*IN_DIM + n];
    }
    __syncthreads();

    int wave = t >> 5, lane = t & 31;
    int tile = blockIdx.x * 8 + wave;
    int lrow = lane & 15, khalf = lane >> 4;
    int grow = tile*16 + lrow;                 // row in [E*2)
    int edge = grow >> 1, grp = grow & 1;
    int sN = src[edge], dN = dst[edge];
    int drow = dN*2 + grp;                     // gather/scatter row for Q_h / output

    const float* ep = e + (size_t)grow * IN_DIM;
    const float* hs = h + (size_t)(sN*2 + grp) * IN_DIM;
    const float* hd = h + (size_t)(dN*2 + grp) * IN_DIM;

    v16bf Ad[2], Ae[2];
    #pragma unroll
    for (int ks = 0; ks < 2; ++ks) {
        int kb = ks*32 + khalf*16;
        fill8_diff(Ad[ks], 0, hs, hd, kb);
        fill8_diff(Ad[ks], 8, hs, hd, kb + 8);
        fill8(Ae[ks], 0, ep, kb);
        fill8(Ae[ks], 8, ep, kb + 8);
    }

    int n = lane & 15, mh = lane >> 4;
    // dst row for each matrix row this lane's accumulator VGPRs hold (hoisted: reused by all heads)
    int qrowA[8];
    #pragma unroll
    for (int j = 0; j < 8; ++j) qrowA[j] = __shfl(drow, mh*8 + j, 32);

    #pragma unroll
    for (int nt = 0; nt < 4; ++nt) {           // nt == head
        v8f ck = {}, cv = {}, cr = {};
        #pragma unroll
        for (int ks = 0; ks < 2; ++ks) {
            v16bf bk = *(const v16bf*)&sB[0][ks][nt][lane][0];
            v16bf bv = *(const v16bf*)&sB[1][ks][nt][lane][0];
            v16bf bb = *(const v16bf*)&sB[2][ks][nt][lane][0];
            ck = wmma_bf16(Ad[ks], bk, ck);
            cv = wmma_bf16(Ad[ks], bv, cv);
            cr = wmma_bf16(Ae[ks], bb, cr);
        }
        #pragma unroll
        for (int j = 0; j < 8; ++j) {
            int qrow = qrowA[j];
            size_t qidx = (size_t)qrow*IN_DIM + nt*16 + n;
            float q  = Qh[qidx];
            float qn = Qn[(size_t)qrow*4 + nt];           // precomputed ||Q_h[dst,head]||
            float kx = ck[j];
            float sk = hsum16(kx*kx);                     // ||k_diff[row, head]||^2
            float scale = fast_sqrt(sk) * qn + 1e-6f;
            float score = q * kx * scale + cr[j];
            float msg   = score * cv[j];
            atomicMax(&outEnc[qidx], f2o(msg));           // segment max over dst
        }
    }
}

// ---------------- Kernel 3: decode + zero-degree -> 0 (4 elems/thread)
__global__ void __launch_bounds__(256)
finalize_kernel(float* __restrict__ out, int nElems4) {
    int i = blockIdx.x * 256 + threadIdx.x;
    if (i < nElems4) {
        uint4 u = ((uint4*)out)[i];
        float4 f;
        f.x = (u.x == ENC_NEG_INF) ? 0.0f : o2f(u.x);
        f.y = (u.y == ENC_NEG_INF) ? 0.0f : o2f(u.y);
        f.z = (u.z == ENC_NEG_INF) ? 0.0f : o2f(u.z);
        f.w = (u.w == ENC_NEG_INF) ? 0.0f : o2f(u.w);
        ((float4*)out)[i] = f;
    }
}

extern "C" void kernel_launch(void* const* d_in, const int* in_sizes, int n_in,
                              void* d_out, int out_size, void* d_ws, size_t ws_size,
                              hipStream_t stream) {
    const float* h  = (const float*)d_in[0];
    const float* e  = (const float*)d_in[1];
    const float* Wq = (const float*)d_in[2];
    const float* Wk = (const float*)d_in[3];
    const float* Wv = (const float*)d_in[4];
    const float* Wr = (const float*)d_in[5];
    const int*   src = (const int*)d_in[6];
    const int*   dst = (const int*)d_in[7];

    int nRowsNode = in_sizes[0] / IN_DIM;     // 50000 (= N_NODES*2)
    int nRowsEdge = in_sizes[1] / IN_DIM;     // 800000 (= N_EDGES*2)
    int nTilesA   = nRowsNode / 16;           // 3125
    int nTilesE   = nRowsEdge / 16;           // 50000

    float*    Qh     = (float*)d_ws;                             // [nRowsNode, 64] f32
    float*    Qn     = (float*)d_ws + (size_t)nRowsNode*IN_DIM;  // [nRowsNode, 4]  f32
    unsigned* outEnc = (unsigned*)d_out;

    proj_q_kernel<<<nTilesA, 32, 0, stream>>>(h, Wq, Qh, Qn, outEnc);
    edge_kernel<<<nTilesE / 8, 256, 0, stream>>>(h, e, Wk, Wv, Wr, src, dst, Qh, Qn, outEnc);
    int nElems4 = out_size / 4;
    finalize_kernel<<<(nElems4 + 255) / 256, 256, 0, stream>>>((float*)d_out, nElems4);
}
